// CustomizingAttention_31619549233728
// MI455X (gfx1250) — matrix operand order repaired and checked by reading, not compiled, not run on hardware
//
#include <hip/hip_runtime.h>

// ---------------------------------------------------------------------------
// MI455X (gfx1250) location-sensitive attention, bf16 WMMA (fp32 accumulate).
// HIDDEN=1024 HEADS=8 DIM=128 B=32 QLEN=128 VLEN=2048
// ---------------------------------------------------------------------------

typedef __attribute__((ext_vector_type(16))) __bf16 v16bf;
typedef __attribute__((ext_vector_type(8)))  __bf16 v8bf;
typedef __attribute__((ext_vector_type(8)))  float  v8f;

#define HIDDEN 1024
#define HEADS  8
#define DIM    128
#define BB     32
#define QLEN   128
#define VLEN   2048

// workspace layout (bytes)
#define LOCW_OFF 0u
#define QH_OFF   4096ull                               // 256*128*128 bf16 = 8 MB
#define VH_OFF   (QH_OFF + 8388608ull)                 // 256*2048*128 bf16 = 128 MB
#define ATTN_OFF (VH_OFF + 134217728ull)               // 256*128*2048 bf16 = 128 MB
#define OUT_ELEMS 4194304ull                           // B*QLEN*HIDDEN floats

// Hardware bf16 convert (RNE fptrunc; backend selects v_cvt_*bf16_f32).
static __device__ __forceinline__ __bf16 f2bf(float f) { return (__bf16)f; }

static __device__ __forceinline__ v16bf cat8(v8bf lo, v8bf hi) {
  return __builtin_shufflevector(lo, hi, 0,1,2,3,4,5,6,7,8,9,10,11,12,13,14,15);
}

static __device__ __forceinline__ v8f wmma_bf16(v16bf a, v16bf b, v8f c) {
  return __builtin_amdgcn_wmma_f32_16x16x32_bf16(false, a, false, b, (short)0, c, false, false);
}

// ---------------------------------------------------------------------------
// Kernel 0: collapse conv(10,1,3)+conv_b+Wloc(128,10) -> WW[128][3], biasLoc[128]
// loc_energy[b,v,h,d] = biasLoc[d] + sum_k WW[d][k]*last_attn[b*8+h, v+k-1]
// ---------------------------------------------------------------------------
__global__ void prep_kernel(const float* __restrict__ Wloc,
                            const float* __restrict__ conv_w,
                            const float* __restrict__ conv_b,
                            float4* __restrict__ locw) {
  int d = threadIdx.x;
  if (d < DIM) {
    float w0 = 0.f, w1 = 0.f, w2 = 0.f, bl = 0.f;
#pragma unroll
    for (int c = 0; c < 10; ++c) {
      float wl = Wloc[d * 10 + c];
      w0 += wl * conv_w[c * 3 + 0];
      w1 += wl * conv_w[c * 3 + 1];
      w2 += wl * conv_w[c * 3 + 2];
      bl += wl * conv_b[c];
    }
    locw[d] = make_float4(w0, w1, w2, bl);
  }
}

// ---------------------------------------------------------------------------
// Kernel 1: projection GEMM  dst = X(Mx1024) @ W^T(1024x1024) + addv [+ loc]
// Block: 256 threads (8 waves), tile 64 rows x 256 cols, K staged 32 in LDS,
// double-buffered so global loads of chunk k+1 overlap WMMA on chunk k.
// mode 0: qp -> qh[h*32+b][ql][d] bf16   mode 1: vp -> vh[h*32+b][v][d] bf16
// ---------------------------------------------------------------------------
__global__ __launch_bounds__(256) void
proj_kernel(const float* __restrict__ X, const float* __restrict__ W,
            const float* __restrict__ addv, const float* __restrict__ la,
            const float4* __restrict__ locw, __bf16* __restrict__ dst, int mode) {
  __shared__ __align__(16) __bf16 As[2][64 * 40];   // [row][k], pad 8 halves
  __shared__ __align__(16) __bf16 Bs[2][256 * 40];  // [o][k],  pad 8 halves

  const int t = threadIdx.x;
  const int lane = t & 31, w = t >> 5;
  const int mw = w & 3;          // m-subtile (16 rows each)
  const int nw = w >> 2;         // 0..1 : 128-col half
  const int g = lane >> 4, l15 = lane & 15;
  const int r0 = blockIdx.y * 64;
  const int o0 = blockIdx.x * 256;

  v8f acc[8];
#pragma unroll
  for (int ct = 0; ct < 8; ++ct) acc[ct] = (v8f){0.f,0.f,0.f,0.f,0.f,0.f,0.f,0.f};

  const int arow = t >> 2, akb = (t & 3) * 8;
  const float* aSrcBase = X + (size_t)(r0 + arow) * HIDDEN + akb;
  const float* bSrcBase = W + (size_t)(o0 + t) * HIDDEN;

  // ---- prologue: stage chunk 0 ----
  {
    const float4* sa = (const float4*)(aSrcBase);
    float4 x0 = sa[0], x1 = sa[1];
    __bf16* dp = As[0] + arow * 40 + akb;
    dp[0]=f2bf(x0.x); dp[1]=f2bf(x0.y); dp[2]=f2bf(x0.z); dp[3]=f2bf(x0.w);
    dp[4]=f2bf(x1.x); dp[5]=f2bf(x1.y); dp[6]=f2bf(x1.z); dp[7]=f2bf(x1.w);
    const float4* sb = (const float4*)(bSrcBase);
    __bf16* bp = Bs[0] + t * 40;
#pragma unroll
    for (int i = 0; i < 8; ++i) {
      float4 x = sb[i];
      bp[i*4+0]=f2bf(x.x); bp[i*4+1]=f2bf(x.y); bp[i*4+2]=f2bf(x.z); bp[i*4+3]=f2bf(x.w);
    }
  }
  __syncthreads();

  int cur = 0;
  for (int k0 = 0; k0 < HIDDEN; k0 += 32) {
    const bool have_next = (k0 + 32 < HIDDEN);
    float4 na0, na1, nb[8];
    if (have_next) {
      const float4* sa = (const float4*)(aSrcBase + k0 + 32);
      na0 = sa[0]; na1 = sa[1];
      const float4* sb = (const float4*)(bSrcBase + k0 + 32);
#pragma unroll
      for (int i = 0; i < 8; ++i) nb[i] = sb[i];
    }

    // A fragment (16x32): lane holds row l15, K = g*16 + j (contiguous)
    const __bf16* ap = As[cur] + (mw * 16 + l15) * 40 + 16 * g;
    v16bf afrag = cat8(*(const v8bf*)ap, *(const v8bf*)(ap + 8));

#pragma unroll
    for (int ct = 0; ct < 8; ++ct) {
      // B fragment (32x16): lane holds col l15, K = g*16 + j (contiguous)
      const __bf16* bpf = Bs[cur] + (nw * 128 + ct * 16 + l15) * 40 + g * 16;
      v16bf bfrag = cat8(*(const v8bf*)bpf, *(const v8bf*)(bpf + 8));
      acc[ct] = wmma_bf16(afrag, bfrag, acc[ct]);
    }

    if (have_next) {
      const int nxt = cur ^ 1;
      __bf16* dp = As[nxt] + arow * 40 + akb;
      dp[0]=f2bf(na0.x); dp[1]=f2bf(na0.y); dp[2]=f2bf(na0.z); dp[3]=f2bf(na0.w);
      dp[4]=f2bf(na1.x); dp[5]=f2bf(na1.y); dp[6]=f2bf(na1.z); dp[7]=f2bf(na1.w);
      __bf16* bp = Bs[nxt] + t * 40;
#pragma unroll
      for (int i = 0; i < 8; ++i) {
        float4 x = nb[i];
        bp[i*4+0]=f2bf(x.x); bp[i*4+1]=f2bf(x.y); bp[i*4+2]=f2bf(x.z); bp[i*4+3]=f2bf(x.w);
      }
      __syncthreads();
      cur = nxt;
    }
  }

  // epilogue
#pragma unroll
  for (int ct = 0; ct < 8; ++ct) {
    const int o = o0 + nw * 128 + ct * 16 + l15;
    const int h = o >> 7, dd = o & 127;
    const float av = addv[o];
    float4 lw = make_float4(0.f, 0.f, 0.f, 0.f);
    if (mode == 1) lw = locw[dd];
#pragma unroll
    for (int r = 0; r < 8; ++r) {
      const int row = r0 + mw * 16 + g * 8 + r;
      float val = acc[ct][r] + av;
      if (mode == 1) {
        const int b = row >> 11, v = row & (VLEN - 1);
        const float* ap2 = la + (size_t)(b * HEADS + h) * VLEN;
        float a0 = (v > 0)        ? ap2[v - 1] : 0.f;
        float a1 = ap2[v];
        float a2 = (v < VLEN - 1) ? ap2[v + 1] : 0.f;
        val += lw.w + lw.x * a0 + lw.y * a1 + lw.z * a2;
        dst[((size_t)(h * BB + b) * VLEN + v) * DIM + dd] = f2bf(val);
      } else {
        const int b = row >> 7, ql = row & (QLEN - 1);
        dst[((size_t)(h * BB + b) * QLEN + ql) * DIM + dd] = f2bf(val);
      }
    }
  }
}

// ---------------------------------------------------------------------------
// Kernel 2: score = qh·vh^T / sqrt(128), softmax over v, write fp32 attn to
// d_out and bf16 attn to ws.  Grid: 2048 blocks (n=bid>>3, qtile=bid&7),
// 512 threads = 16 waves; wave w owns v-cols [w*128, w*128+128).
// ---------------------------------------------------------------------------
__global__ __launch_bounds__(512) void
attn_kernel(const __bf16* __restrict__ qh, const __bf16* __restrict__ vh,
            float* __restrict__ attn_out, __bf16* __restrict__ attn_bf) {
  __shared__ float red[16 * 16];   // [wave][row]

  const int t = threadIdx.x, lane = t & 31, w = t >> 5;       // 16 waves
  const int n = blockIdx.x >> 3, q0 = (blockIdx.x & 7) * 16;
  const int g = lane >> 4, l15 = lane & 15;
  const int v0 = w * 128;

  // A fragments for full K=128
  v16bf afr[4];
  const __bf16* qb = qh + ((size_t)n * QLEN + q0 + l15) * DIM;
#pragma unroll
  for (int kk = 0; kk < 4; ++kk) {
    const __bf16* p = qb + kk * 32 + 16 * g;
    afr[kk] = cat8(*(const v8bf*)p, *(const v8bf*)(p + 8));
  }

  v8f acc[8];
#pragma unroll
  for (int ct = 0; ct < 8; ++ct) {
    acc[ct] = (v8f){0.f,0.f,0.f,0.f,0.f,0.f,0.f,0.f};
    const __bf16* vb = vh + ((size_t)n * VLEN + v0 + ct * 16 + l15) * DIM;
#pragma unroll
    for (int kk = 0; kk < 4; ++kk) {
      v16bf bfr = *(const v16bf*)(vb + kk * 32 + g * 16);  // 32B-aligned contiguous
      acc[ct] = wmma_bf16(afr[kk], bfr, acc[ct]);
    }
  }

  const float sc = 0.08838834764831845f;  // 1/sqrt(128)
#pragma unroll
  for (int ct = 0; ct < 8; ++ct)
#pragma unroll
    for (int r = 0; r < 8; ++r) acc[ct][r] *= sc;

  // --- row max: intra-wave over 8 col-tiles + 16 lanes of half-wave ---
  float rmax[8];
#pragma unroll
  for (int r = 0; r < 8; ++r) {
    float m = acc[0][r];
#pragma unroll
    for (int ct = 1; ct < 8; ++ct) m = fmaxf(m, acc[ct][r]);
    m = fmaxf(m, __shfl_xor(m, 1));
    m = fmaxf(m, __shfl_xor(m, 2));
    m = fmaxf(m, __shfl_xor(m, 4));
    m = fmaxf(m, __shfl_xor(m, 8));
    rmax[r] = m;
  }
  if (l15 == 0) {
#pragma unroll
    for (int r = 0; r < 8; ++r) red[w * 16 + g * 8 + r] = rmax[r];
  }
  __syncthreads();
  float gmax[8];
#pragma unroll
  for (int r = 0; r < 8; ++r) {
    float m = -3.0e38f;
    for (int w2 = 0; w2 < 16; ++w2) m = fmaxf(m, red[w2 * 16 + g * 8 + r]);
    gmax[r] = m;
  }
  __syncthreads();

  // --- exp + row sum ---
  float rsum[8];
#pragma unroll
  for (int r = 0; r < 8; ++r) rsum[r] = 0.f;
#pragma unroll
  for (int ct = 0; ct < 8; ++ct)
#pragma unroll
    for (int r = 0; r < 8; ++r) {
      float p = __expf(acc[ct][r] - gmax[r]);
      acc[ct][r] = p;
      rsum[r] += p;
    }
#pragma unroll
  for (int r = 0; r < 8; ++r) {
    float s = rsum[r];
    s += __shfl_xor(s, 1); s += __shfl_xor(s, 2);
    s += __shfl_xor(s, 4); s += __shfl_xor(s, 8);
    rsum[r] = s;
  }
  if (l15 == 0) {
#pragma unroll
    for (int r = 0; r < 8; ++r) red[w * 16 + g * 8 + r] = rsum[r];
  }
  __syncthreads();
  float rinv[8];
#pragma unroll
  for (int r = 0; r < 8; ++r) {
    float s = 0.f;
    for (int w2 = 0; w2 < 16; ++w2) s += red[w2 * 16 + g * 8 + r];
    rinv[r] = 1.0f / s;
  }

  // --- normalize + store (fp32 to d_out, bf16 to ws) ---
  const size_t base = ((size_t)n * QLEN + q0) * VLEN;
#pragma unroll
  for (int ct = 0; ct < 8; ++ct) {
    const int col = v0 + ct * 16 + l15;
#pragma unroll
    for (int r = 0; r < 8; ++r) {
      const int row = g * 8 + r;
      float p = acc[ct][r] * rinv[r];
      attn_out[base + (size_t)row * VLEN + col] = p;
      attn_bf [base + (size_t)row * VLEN + col] = f2bf(p);
    }
  }
}

// ---------------------------------------------------------------------------
// Kernel 3: out = attn(128x2048) @ vh(2048x128) per n; scatter into
// out[b][ql][h*128+d].  Grid 512 (n = bid>>1, q-half = bid&1), 128 thr/4 waves.
// Double-buffered 32x128 vh tile in LDS.
// ---------------------------------------------------------------------------
__global__ __launch_bounds__(128) void
out_kernel(const __bf16* __restrict__ attn_bf, const __bf16* __restrict__ vh,
           float* __restrict__ outp) {
  __shared__ __align__(16) __bf16 Bs[2][32 * 136];  // vh tile [k=32][d=128], pad 8

  const int t = threadIdx.x, lane = t & 31, w = t >> 5;  // 4 waves
  const int n = blockIdx.x >> 1;
  const int q0 = (blockIdx.x & 1) * 64 + w * 16;
  const int g = lane >> 4, l15 = lane & 15;
  const int h = n >> 5, b = n & 31;

  v8f acc[8];
#pragma unroll
  for (int dt = 0; dt < 8; ++dt) acc[dt] = (v8f){0.f,0.f,0.f,0.f,0.f,0.f,0.f,0.f};

  const __bf16* arow = attn_bf + ((size_t)n * QLEN + q0 + l15) * VLEN;
  const int fvl = t >> 2, fdc = (t & 3) * 32;
  const v8bf* vSrcBase = (const v8bf*)(vh + ((size_t)n * VLEN + fvl) * DIM + fdc);
  const int vSrcStride = (32 * DIM) / 8;   // v8bf elements per 32-row step

  // prologue: stage tile v=0
  {
    v8bf x0 = vSrcBase[0], x1 = vSrcBase[1], x2 = vSrcBase[2], x3 = vSrcBase[3];
    __bf16* dp = Bs[0] + fvl * 136 + fdc;
    *(v8bf*)(dp + 0) = x0; *(v8bf*)(dp + 8)  = x1;
    *(v8bf*)(dp + 16) = x2; *(v8bf*)(dp + 24) = x3;
  }
  __syncthreads();

  int cur = 0;
  for (int v = 0; v < VLEN; v += 32) {
    const bool have_next = (v + 32 < VLEN);
    v8bf n0, n1, n2, n3;
    if (have_next) {
      const v8bf* s = vSrcBase + (size_t)((v >> 5) + 1) * vSrcStride;
      n0 = s[0]; n1 = s[1]; n2 = s[2]; n3 = s[3];
    }

    const __bf16* ap = arow + v + 16 * g;
    v16bf afrag = cat8(*(const v8bf*)ap, *(const v8bf*)(ap + 8));

#pragma unroll
    for (int dt = 0; dt < 8; ++dt) {
      v16bf bfrag;
#pragma unroll
      for (int j = 0; j < 16; ++j)
        bfrag[j] = Bs[cur][(g * 16 + j) * 136 + dt * 16 + l15];
      acc[dt] = wmma_bf16(afrag, bfrag, acc[dt]);
    }

    if (have_next) {
      const int nxt = cur ^ 1;
      __bf16* dp = Bs[nxt] + fvl * 136 + fdc;
      *(v8bf*)(dp + 0) = n0; *(v8bf*)(dp + 8)  = n1;
      *(v8bf*)(dp + 16) = n2; *(v8bf*)(dp + 24) = n3;
      __syncthreads();
      cur = nxt;
    }
  }

  const size_t ob = (size_t)b * QLEN * HIDDEN + (size_t)h * DIM;
#pragma unroll
  for (int dt = 0; dt < 8; ++dt) {
    const int d = dt * 16 + l15;
#pragma unroll
    for (int r = 0; r < 8; ++r) {
      const int ql = q0 + g * 8 + r;
      outp[ob + (size_t)ql * HIDDEN + d] = acc[dt][r];
    }
  }
}

// ---------------------------------------------------------------------------
extern "C" void kernel_launch(void* const* d_in, const int* in_sizes, int n_in,
                              void* d_out, int out_size, void* d_ws, size_t ws_size,
                              hipStream_t stream) {
  const float* q         = (const float*)d_in[0];
  const float* v         = (const float*)d_in[1];
  const float* last_attn = (const float*)d_in[2];
  const float* conv_w    = (const float*)d_in[3];
  const float* conv_b    = (const float*)d_in[4];
  const float* Wq        = (const float*)d_in[5];
  const float* bq        = (const float*)d_in[6];
  const float* Wv        = (const float*)d_in[7];
  const float* Wloc      = (const float*)d_in[8];
  const float* bias      = (const float*)d_in[9];

  char* ws = (char*)d_ws;
  float4* locw    = (float4*)(ws + LOCW_OFF);
  __bf16* qh      = (__bf16*)(ws + QH_OFF);
  __bf16* vh      = (__bf16*)(ws + VH_OFF);
  __bf16* attn_bf = (__bf16*)(ws + ATTN_OFF);

  float* outp     = (float*)d_out;
  float* attn_out = outp + OUT_ELEMS;

  prep_kernel<<<1, 128, 0, stream>>>(Wloc, conv_w, conv_b, locw);

  dim3 gq(4, (BB * QLEN) / 64);   // (4, 64)
  proj_kernel<<<gq, 256, 0, stream>>>(q, Wq, bq, last_attn, locw, qh, 0);

  dim3 gv(4, (BB * VLEN) / 64);   // (4, 1024)
  proj_kernel<<<gv, 256, 0, stream>>>(v, Wv, bias, last_attn, locw, vh, 1);

  attn_kernel<<<HEADS * BB * (QLEN / 16), 512, 0, stream>>>(qh, vh, attn_out, attn_bf);

  out_kernel<<<HEADS * BB * 2, 128, 0, stream>>>(attn_bf, vh, outp);
}
